// CausalSelfAttention_80745385164915
// MI455X (gfx1250) — matrix operand order repaired and checked by reading, not compiled, not run on hardware
//
#include <hip/hip_runtime.h>

typedef __attribute__((ext_vector_type(16))) __bf16 v16bf;
typedef __attribute__((ext_vector_type(8)))  float  v8f;

union AB16 {
  v16bf        v;
  unsigned int u[8];
  __bf16       h[16];
};

// A-fragment K index for 16-bit A 16x32 layout (ISA 7.12.2)
__device__ __forceinline__ int kk_a(int i, int half) {
  return ((i >> 2) << 4) + (half << 3) + ((i & 3) << 1);
}

__device__ __forceinline__ float rmax16(float v) {
  v = fmaxf(v, __shfl_xor(v, 1, 32));
  v = fmaxf(v, __shfl_xor(v, 2, 32));
  v = fmaxf(v, __shfl_xor(v, 4, 32));
  v = fmaxf(v, __shfl_xor(v, 8, 32));
  return v;
}
__device__ __forceinline__ float rsum16(float v) {
  v += __shfl_xor(v, 1, 32);
  v += __shfl_xor(v, 2, 32);
  v += __shfl_xor(v, 4, 32);
  v += __shfl_xor(v, 8, 32);
  return v;
}

// ---------------------------------------------------------------------------
// Kernel 1: qkv = x @ W_attn + b_attn  -> scatter into Q[B,H,T,d], K[B,H,T,d],
// Vt[B,H,d,T] as bf16. M=4096, K=1024, N=3072. Block = 8 waves, wave = 32x64.
// The 64-wide N tile maps to exactly one of {Q,K,V} and one head (64 | 1024),
// so `which` and `h` are block-uniform -> branchless store loops.
// ---------------------------------------------------------------------------
__global__ __launch_bounds__(256) void qkv_gemm_kernel(
    const float* __restrict__ x, const float* __restrict__ W,
    const float* __restrict__ bias,
    __bf16* __restrict__ qb, __bf16* __restrict__ kb, __bf16* __restrict__ vt)
{
  __shared__ __align__(16) __bf16 wt[64 * 32]; // W tile, transposed: [ncol][krow]
  const int ldn = 3072;
  const int tid  = threadIdx.x;
  const int w    = tid >> 5;
  const int lane = tid & 31;
  const int half = lane >> 4;
  const int l16  = lane & 15;
  const int n0 = blockIdx.x * 64;
  const int m0 = blockIdx.y * 256 + w * 32;

  v8f acc[2][4] = {};

  for (int k0 = 0; k0 < 1024; k0 += 32) {
    __syncthreads();
    // Stage W (32K x 64N) tile into LDS, bf16, transposed for pair reads.
#pragma unroll
    for (int j = 0; j < 8; ++j) {
      int g  = tid + 256 * j;       // 0..2047, coalesced over columns
      int kr = g >> 6, nc = g & 63;
      wt[nc * 32 + kr] = (__bf16)W[(size_t)(k0 + kr) * ldn + n0 + nc];
    }
    __syncthreads();

    // A fragments: two 16x32 bf16 tiles converted from fp32 x rows
    AB16 a[2];
#pragma unroll
    for (int rb = 0; rb < 2; ++rb) {
      const float* p = x + (size_t)(m0 + rb * 16 + l16) * 1024 + k0;
#pragma unroll
      for (int i = 0; i < 8; ++i) {
        int kk = kk_a(i, half);
        a[rb].h[2 * i]     = (__bf16)p[kk];
        a[rb].h[2 * i + 1] = (__bf16)p[kk + 1];
      }
    }
#pragma unroll
    for (int ct = 0; ct < 4; ++ct) {
      AB16 b;
#pragma unroll
      for (int i = 0; i < 8; ++i)
        b.u[i] = *(const unsigned int*)(&wt[(ct * 16 + l16) * 32 + kk_a(i, half)]);
#pragma unroll
      for (int rb = 0; rb < 2; ++rb)
        acc[rb][ct] = __builtin_amdgcn_wmma_f32_16x16x32_bf16(
            false, a[rb].v, false, b.v, (short)0, acc[rb][ct], false, false);
    }
  }

  // ---- Epilogue: block-uniform routing ----
  const int which = n0 >> 10;           // 0=q 1=k 2=v (uniform per block)
  const int hh    = (n0 & 1023) >> 6;   // head index   (uniform per block)

  float bv[4];
#pragma unroll
  for (int ct = 0; ct < 4; ++ct) bv[ct] = bias[n0 + ct * 16 + l16];

  if (which == 2) {
    // Vt[B,H,d,T]: element -> vt[((bb*16+hh)*64 + dd)*2048 + t]
#pragma unroll
    for (int rb = 0; rb < 2; ++rb)
#pragma unroll
      for (int i = 0; i < 8; ++i) {
        int r  = m0 + rb * 16 + half * 8 + i;
        int bb = r >> 11, t = r & 2047;
        __bf16* base = vt + ((size_t)(bb * 16 + hh) * 64) * 2048 + t;
#pragma unroll
        for (int ct = 0; ct < 4; ++ct) {
          int dd = ct * 16 + l16;
          base[(size_t)dd * 2048] = (__bf16)(acc[rb][ct][i] + bv[ct]);
        }
      }
  } else {
    __bf16* dst = (which == 0) ? qb : kb;
    // [B,H,T,d]: element -> dst[((bb*16+hh)*2048 + t)*64 + dd]
#pragma unroll
    for (int rb = 0; rb < 2; ++rb)
#pragma unroll
      for (int i = 0; i < 8; ++i) {
        int r  = m0 + rb * 16 + half * 8 + i;
        int bb = r >> 11, t = r & 2047;
        __bf16* base = dst + ((size_t)(bb * 16 + hh) * 2048 + t) * 64;
#pragma unroll
        for (int ct = 0; ct < 4; ++ct)
          base[ct * 16 + l16] = (__bf16)(acc[rb][ct][i] + bv[ct]);
      }
  }
}

// ---------------------------------------------------------------------------
// Kernel 2: flash-style causal attention. grid = (B*H, T/128), 8 waves/block,
// each wave owns 16 queries, streams 32-key blocks with online softmax.
// No workgroup barriers (waves have different causal trip counts); the P
// D->A layout transpose goes through a per-wave private LDS patch.
// ---------------------------------------------------------------------------
__global__ __launch_bounds__(256) void attn_kernel(
    const __bf16* __restrict__ qg, const __bf16* __restrict__ kg,
    const __bf16* __restrict__ vtg, __bf16* __restrict__ y)
{
  __shared__ __align__(16) __bf16 psm[8 * 16 * 32]; // per-wave 16x32 P patch
  const int tid  = threadIdx.x;
  const int w    = tid >> 5;
  const int lane = tid & 31;
  const int half = lane >> 4;
  const int l16  = lane & 15;
  const int bh = blockIdx.x;
  const int qw = blockIdx.y * 128 + w * 16;

  const __bf16* qh = qg  + (size_t)bh * 2048 * 64;
  const __bf16* kh = kg  + (size_t)bh * 2048 * 64;
  const __bf16* vh = vtg + (size_t)bh * 64 * 2048;
  __bf16* my = psm + w * 512;

  // Q fragments (reused across all key blocks): 16 queries x 64 d
  AB16 qa[2];
#pragma unroll
  for (int s = 0; s < 2; ++s) {
    const __bf16* p = qh + (size_t)(qw + l16) * 64 + s * 32;
#pragma unroll
    for (int i = 0; i < 8; ++i)
      qa[s].u[i] = *(const unsigned int*)(p + kk_a(i, half));
  }

  v8f o[4] = {};
  float mrow[8], lrow[8];
#pragma unroll
  for (int i = 0; i < 8; ++i) { mrow[i] = -1.0e30f; lrow[i] = 0.0f; }

  const int nkb = (qw + 15) / 32 + 1;   // causal: keys <= qw+15
  for (int kb = 0; kb < nkb; ++kb) {
    const int kbase = kb * 32;

    // K^T B-fragments: 2 key tiles x 2 d-steps
    AB16 bk[2][2];
#pragma unroll
    for (int kt = 0; kt < 2; ++kt)
#pragma unroll
      for (int s = 0; s < 2; ++s) {
        const __bf16* p = kh + (size_t)(kbase + kt * 16 + l16) * 64 + s * 32;
#pragma unroll
        for (int i = 0; i < 8; ++i)
          bk[kt][s].u[i] = *(const unsigned int*)(p + kk_a(i, half));
      }

    // S = Q K^T (16 x 32), fp32 accum
    v8f sf[2] = {};
#pragma unroll
    for (int kt = 0; kt < 2; ++kt) {
      sf[kt] = __builtin_amdgcn_wmma_f32_16x16x32_bf16(
          false, qa[0].v, false, bk[kt][0].v, (short)0, sf[kt], false, false);
      sf[kt] = __builtin_amdgcn_wmma_f32_16x16x32_bf16(
          false, qa[1].v, false, bk[kt][1].v, (short)0, sf[kt], false, false);
    }

    // scale + causal mask + online softmax update, per D-frag row
#pragma unroll
    for (int i = 0; i < 8; ++i) {
      int qi = qw + half * 8 + i;
      float s0 = sf[0][i] * 0.125f;
      float s1 = sf[1][i] * 0.125f;
      if (kbase + l16      > qi) s0 = -1.0e30f;
      if (kbase + 16 + l16 > qi) s1 = -1.0e30f;
      float rm   = rmax16(fmaxf(s0, s1));
      float mnew = fmaxf(mrow[i], rm);
      float alpha = __expf(mrow[i] - mnew);
      float p0 = __expf(s0 - mnew);
      float p1 = __expf(s1 - mnew);
      float rs = rsum16(p0 + p1);
      lrow[i] = lrow[i] * alpha + rs;
      mrow[i] = mnew;
#pragma unroll
      for (int ct = 0; ct < 4; ++ct) o[ct][i] *= alpha;
      // stash P (bf16) in this wave's LDS patch, row-major 16x32
      my[(half * 8 + i) * 32 + l16]      = (__bf16)p0;
      my[(half * 8 + i) * 32 + 16 + l16] = (__bf16)p1;
    }

    asm volatile("s_wait_dscnt 0" ::: "memory");

    // Reload P as an A fragment (16 x 32)
    AB16 pa;
#pragma unroll
    for (int i = 0; i < 8; ++i)
      pa.u[i] = *(const unsigned int*)(my + l16 * 32 + kk_a(i, half));

    // O += P @ V : V B-fragments from transposed Vt (contiguous key pairs)
#pragma unroll
    for (int ct = 0; ct < 4; ++ct) {
      AB16 bvf;
      const __bf16* p = vh + (size_t)(ct * 16 + l16) * 2048 + kbase;
#pragma unroll
      for (int i = 0; i < 8; ++i)
        bvf.u[i] = *(const unsigned int*)(p + kk_a(i, half));
      o[ct] = __builtin_amdgcn_wmma_f32_16x16x32_bf16(
          false, pa.v, false, bvf.v, (short)0, o[ct], false, false);
    }
  }

  // Epilogue: normalize rows, write y[B,T,C] as bf16
  const int bb = bh >> 4;
  const int h  = bh & 15;
#pragma unroll
  for (int i = 0; i < 8; ++i) {
    float inv = 1.0f / lrow[i];
    int qi = qw + half * 8 + i;
    __bf16* base = y + ((size_t)bb * 2048 + qi) * 1024 + h * 64 + l16;
#pragma unroll
    for (int ct = 0; ct < 4; ++ct)
      base[ct * 16] = (__bf16)(o[ct][i] * inv);
  }
}

// ---------------------------------------------------------------------------
// Kernel 3: out = y @ W_proj + b_proj (fp32 out). M=4096, K=1024, N=1024.
// ---------------------------------------------------------------------------
__global__ __launch_bounds__(256) void proj_gemm_kernel(
    const __bf16* __restrict__ y, const float* __restrict__ W,
    const float* __restrict__ bias, float* __restrict__ out)
{
  __shared__ __align__(16) __bf16 wt[64 * 32];
  const int ldn = 1024;
  const int tid  = threadIdx.x;
  const int w    = tid >> 5;
  const int lane = tid & 31;
  const int half = lane >> 4;
  const int l16  = lane & 15;
  const int n0 = blockIdx.x * 64;
  const int m0 = blockIdx.y * 256 + w * 32;

  v8f acc[2][4] = {};

  for (int k0 = 0; k0 < 1024; k0 += 32) {
    __syncthreads();
#pragma unroll
    for (int j = 0; j < 8; ++j) {
      int g  = tid + 256 * j;
      int kr = g >> 6, nc = g & 63;
      wt[nc * 32 + kr] = (__bf16)W[(size_t)(k0 + kr) * ldn + n0 + nc];
    }
    __syncthreads();

    AB16 a[2];
#pragma unroll
    for (int rb = 0; rb < 2; ++rb) {
      const __bf16* p = y + (size_t)(m0 + rb * 16 + l16) * 1024 + k0;
#pragma unroll
      for (int i = 0; i < 8; ++i)
        a[rb].u[i] = *(const unsigned int*)(p + kk_a(i, half));
    }
#pragma unroll
    for (int ct = 0; ct < 4; ++ct) {
      AB16 b;
#pragma unroll
      for (int i = 0; i < 8; ++i)
        b.u[i] = *(const unsigned int*)(&wt[(ct * 16 + l16) * 32 + kk_a(i, half)]);
#pragma unroll
      for (int rb = 0; rb < 2; ++rb)
        acc[rb][ct] = __builtin_amdgcn_wmma_f32_16x16x32_bf16(
            false, a[rb].v, false, b.v, (short)0, acc[rb][ct], false, false);
    }
  }

  float bvv[4];
#pragma unroll
  for (int ct = 0; ct < 4; ++ct) bvv[ct] = bias[n0 + ct * 16 + l16];

#pragma unroll
  for (int rb = 0; rb < 2; ++rb)
#pragma unroll
    for (int i = 0; i < 8; ++i) {
      int r = m0 + rb * 16 + half * 8 + i;
      float* base = out + (size_t)r * 1024 + n0 + l16;
#pragma unroll
      for (int ct = 0; ct < 4; ++ct)
        base[ct * 16] = acc[rb][ct][i] + bvv[ct];
    }
}

// ---------------------------------------------------------------------------
extern "C" void kernel_launch(void* const* d_in, const int* in_sizes, int n_in,
                              void* d_out, int out_size, void* d_ws, size_t ws_size,
                              hipStream_t stream) {
  (void)in_sizes; (void)n_in; (void)out_size; (void)ws_size;
  const float* x      = (const float*)d_in[0];
  const float* W_attn = (const float*)d_in[1];
  const float* b_attn = (const float*)d_in[2];
  const float* W_proj = (const float*)d_in[3];
  const float* b_proj = (const float*)d_in[4];
  float* out = (float*)d_out;

  char* ws = (char*)d_ws;
  const size_t E = (size_t)2 * 2048 * 1024;   // 4M elements per tensor
  __bf16* q  = (__bf16*)(ws);
  __bf16* k  = (__bf16*)(ws + E * 2);
  __bf16* vt = (__bf16*)(ws + E * 4);
  __bf16* y  = (__bf16*)(ws + E * 6);

  // qkv: M=4096 / (8 waves * 32) = 16 blocks y; N=3072/64 = 48 blocks x
  qkv_gemm_kernel<<<dim3(48, 16), 256, 0, stream>>>(x, W_attn, b_attn, q, k, vt);
  // attention: B*H = 32 heads, T/128 = 16 query blocks
  attn_kernel<<<dim3(32, 16), 256, 0, stream>>>(q, k, vt, y);
  // proj: N=1024/64 = 16 blocks x, M -> 16 blocks y
  proj_gemm_kernel<<<dim3(16, 16), 256, 0, stream>>>(y, W_proj, b_proj, out);
}
